// SCNN_63556926046345
// MI455X (gfx1250) — matrix operand order repaired and checked by reading, not compiled, not run on hardware
//
#include <hip/hip_runtime.h>
#include <hip/hip_bf16.h>

// ---------------------------------------------------------------------------
// SCNN (VGG16 + spatial message passing) for MI455X / gfx1250.
// Conv layers: implicit-GEMM with v_wmma_f32_16x16x32_f16 (f16 in, f32 acc).
// Activations NHWC f16; weights repacked per-call into B-fragment layout.
// Main conv wave tile: 32 pixels x 32 couts (2x2 WMMA subtiles).
// ---------------------------------------------------------------------------

typedef __attribute__((ext_vector_type(16))) _Float16 v16h;
typedef __attribute__((ext_vector_type(8)))  _Float16 v8h;
typedef __attribute__((ext_vector_type(8)))  float    v8f;

#define DIV_UP(a, b) (((a) + (b) - 1) / (b))

// ---------------------------------------------------------------------------
// Weight repack: OIHW fp32 -> f16 B-fragment layout.
// Fragment f = ((tap*nCC + cc)*nNT + nt): 32x16 (KxN) block, 512 halves.
// Within fragment, lane L holds halves [L*16 .. L*16+15]:
//   component c -> K = 8*(L>>4) + (c&7) + 16*(c>>3), N = L&15   (ISA 16-bit B layout).
// ---------------------------------------------------------------------------
__global__ void repack_weights(const float* __restrict__ w, _Float16* __restrict__ dst,
                               int Cin, int Cout, int KH, int KW) {
  int nCC = Cin >> 5, nNT = Cout >> 4, taps = KH * KW;
  long long total = (long long)taps * nCC * nNT * 512;
  long long d = (long long)blockIdx.x * 256 + threadIdx.x;
  if (d >= total) return;
  int c = (int)(d & 15);
  int L = (int)((d >> 4) & 31);
  long long rest = d >> 9;
  int nt = (int)(rest % nNT); rest /= nNT;
  int cc = (int)(rest % nCC);
  int tap = (int)(rest / nCC);
  int cin  = cc * 32 + ((L >> 4) << 3) + (c & 7) + ((c >> 3) << 4);
  int cout = nt * 16 + (L & 15);
  int ky = tap / KW, kx = tap % KW;
  float v = w[(((long long)cout * Cin + cin) * KH + ky) * KW + kx];
  dst[d] = (_Float16)v;
}

// ---------------------------------------------------------------------------
// Layer 1 (Cin=3, not WMMA-friendly): direct conv, NCHW fp32 -> NHWC f16,
// fused BN scale/bias + ReLU. ~3 GFLOP, negligible.
// ---------------------------------------------------------------------------
__global__ void conv1_direct(const float* __restrict__ x, const float* __restrict__ w,
                             const float* __restrict__ s, const float* __restrict__ b,
                             _Float16* __restrict__ y) {
  const int H = 288, W = 800, Co = 64;
  long long idx = (long long)blockIdx.x * 256 + threadIdx.x;
  long long total = 4LL * H * W * Co;
  if (idx >= total) return;
  int co = (int)(idx % Co); long long p = idx / Co;
  int xx = (int)(p % W); p /= W;
  int yy = (int)(p % H); int bb = (int)(p / H);
  float acc = 0.f;
  for (int ci = 0; ci < 3; ++ci)
    for (int ky = 0; ky < 3; ++ky) {
      int iy = yy + ky - 1; if (iy < 0 || iy >= H) continue;
      for (int kx = 0; kx < 3; ++kx) {
        int ix = xx + kx - 1; if (ix < 0 || ix >= W) continue;
        acc += x[((long long)(bb * 3 + ci) * H + iy) * W + ix] *
               w[((co * 3 + ci) * 3 + ky) * 3 + kx];
      }
    }
  float v = acc * s[co] + b[co];
  y[idx] = (_Float16)(v > 0.f ? v : 0.f);
}

// ---------------------------------------------------------------------------
// Generic conv (KHxKW, dilation, pad) via WMMA implicit GEMM.
// X: NHWC f16. Wp: packed fragments. Per wave: 32 pixels x 32 couts
// (2 A-subtiles x 2 B-subtiles = 4 WMMAs per K-step).
// A layout (16-bit A 16x32, wave32): lane L, g=L>>4, row M=L&15,
//   components 0..7  -> K = 8g+c        -> channels cinBase+8g+[0..7]
//   components 8..15 -> K = 16+8g+(c-8) -> channels cinBase+16+8g+[0..7]
// Both chunks are contiguous 16B v8h loads from NHWC memory.
// ---------------------------------------------------------------------------
__global__ __launch_bounds__(256)
void conv_wmma(const _Float16* __restrict__ X, const _Float16* __restrict__ Wp,
               const float* __restrict__ scale, const float* __restrict__ bias,
               _Float16* __restrict__ Y,
               int Bn, int H, int Wd, int Cin, int Cout,
               int KH, int KW, int dil, int pad) {
  int lane = threadIdx.x & 31;
  long long gw = (long long)blockIdx.x * 8 + (threadIdx.x >> 5);
  int nNT2 = Cout >> 5;                         // 32-wide cout tiles
  long long M = (long long)Bn * H * Wd;
  long long nMT2 = (M + 31) >> 5;               // 32-pixel tiles
  if (gw >= nMT2 * nNT2) return;                // wave-uniform exit
  int nt2 = (int)(gw % nNT2);
  long long mt = gw / nNT2;

  int g = lane >> 4;
  long long p0 = mt * 32 + (lane & 15);
  long long p1 = p0 + 16;
  bool pv0 = p0 < M, pv1 = p1 < M;
  long long q0 = pv0 ? p0 : 0, q1 = pv1 ? p1 : 0;
  int ox0 = (int)(q0 % Wd); long long t0 = q0 / Wd;
  int oy0 = (int)(t0 % H);  int bb0 = (int)(t0 / H);
  int ox1 = (int)(q1 % Wd); long long t1 = q1 / Wd;
  int oy1 = (int)(t1 % H);  int bb1 = (int)(t1 / H);
  int nCC = Cin >> 5, nNT = Cout >> 4;
  int taps = KH * KW;

  v8f acc00 = {}; v8f acc01 = {}; v8f acc10 = {}; v8f acc11 = {};
  for (int tap = 0; tap < taps; ++tap) {
    int ky = tap / KW, kx = tap % KW;
    int iy0 = oy0 + ky * dil - pad, ix0 = ox0 + kx * dil - pad;
    int iy1 = oy1 + ky * dil - pad, ix1 = ox1 + kx * dil - pad;
    bool in0 = pv0 && iy0 >= 0 && iy0 < H && ix0 >= 0 && ix0 < Wd;
    bool in1 = pv1 && iy1 >= 0 && iy1 < H && ix1 >= 0 && ix1 < Wd;
    const _Float16* xr0 = X + (((long long)bb0 * H + iy0) * Wd + ix0) * Cin + 8 * g;
    const _Float16* xr1 = X + (((long long)bb1 * H + iy1) * Wd + ix1) * Cin + 8 * g;
    const _Float16* wtap = Wp + (long long)tap * nCC * nNT * 512;
    for (int cc = 0; cc < nCC; ++cc) {
      v16h A0 = {}; v16h A1 = {};
      if (in0) {
        union { v16h v; v8h h[2]; } u;
        u.h[0] = *(const v8h*)(xr0 + cc * 32);
        u.h[1] = *(const v8h*)(xr0 + cc * 32 + 16);
        A0 = u.v;
      }
      if (in1) {
        union { v16h v; v8h h[2]; } u;
        u.h[0] = *(const v8h*)(xr1 + cc * 32);
        u.h[1] = *(const v8h*)(xr1 + cc * 32 + 16);
        A1 = u.v;
      }
      const _Float16* wf = wtap + ((long long)cc * nNT + nt2 * 2) * 512 + lane * 16;
      v16h B0 = *(const v16h*)(wf);
      v16h B1 = *(const v16h*)(wf + 512);
      acc00 = __builtin_amdgcn_wmma_f32_16x16x32_f16(false, A0, false, B0, (short)0, acc00, false, false);
      acc01 = __builtin_amdgcn_wmma_f32_16x16x32_f16(false, A0, false, B1, (short)0, acc01, false, false);
      acc10 = __builtin_amdgcn_wmma_f32_16x16x32_f16(false, A1, false, B0, (short)0, acc10, false, false);
      acc11 = __builtin_amdgcn_wmma_f32_16x16x32_f16(false, A1, false, B1, (short)0, acc11, false, false);
    }
  }
  // Epilogue: D layout -> component el holds M = el + 8g, lane holds N.
  int co0 = nt2 * 32 + (lane & 15);
  int co1 = co0 + 16;
  float s0 = scale[co0], b0 = bias[co0];
  float s1 = scale[co1], b1 = bias[co1];
  for (int el = 0; el < 8; ++el) {
    long long pix0 = mt * 32 + el + 8 * g;
    long long pix1 = pix0 + 16;
    if (pix0 < M) {
      _Float16* yp = Y + pix0 * Cout;
      float a0 = acc00[el] * s0 + b0; if (a0 < 0.f) a0 = 0.f;
      float a1 = acc01[el] * s1 + b1; if (a1 < 0.f) a1 = 0.f;
      yp[co0] = (_Float16)a0;
      yp[co1] = (_Float16)a1;
    }
    if (pix1 < M) {
      _Float16* yp = Y + pix1 * Cout;
      float a0 = acc10[el] * s0 + b0; if (a0 < 0.f) a0 = 0.f;
      float a1 = acc11[el] * s1 + b1; if (a1 < 0.f) a1 = 0.f;
      yp[co0] = (_Float16)a0;
      yp[co1] = (_Float16)a1;
    }
  }
}

// ---------------------------------------------------------------------------
// One SCNN scan step: Y[b,t,:] += relu(conv1d_k9(X)[b,t,:]), strided slices.
// Same WMMA structure, C = 128, 16-pixel x 32-cout wave tile (M is small).
// ---------------------------------------------------------------------------
__global__ __launch_bounds__(256)
void scan_step_wmma(const _Float16* __restrict__ Xp, _Float16* __restrict__ Yp,
                    const _Float16* __restrict__ Wpk,
                    int Bn, int L, long long sB, long long sT, int C) {
  int lane = threadIdx.x & 31;
  long long gw = (long long)blockIdx.x * 8 + (threadIdx.x >> 5);
  int nNT2 = C >> 5;
  long long M = (long long)Bn * L;
  long long nMT = (M + 15) >> 4;
  if (gw >= nMT * nNT2) return;
  int nt2 = (int)(gw % nNT2);
  long long mt = gw / nNT2;

  long long p = mt * 16 + (lane & 15);
  bool pvalid = p < M;
  long long pc = pvalid ? p : 0;
  int tt = (int)(pc % L); int bb = (int)(pc / L);
  int g = lane >> 4;
  int nCC = C >> 5, nNT = C >> 4;

  v8f acc0 = {}; v8f acc1 = {};
  for (int tap = 0; tap < 9; ++tap) {
    int it = tt + tap - 4;
    bool inb = pvalid && it >= 0 && it < L;
    const _Float16* xr = Xp + (long long)bb * sB + (long long)it * sT + 8 * g;
    const _Float16* wtap = Wpk + (long long)tap * nCC * nNT * 512;
    for (int cc = 0; cc < nCC; ++cc) {
      v16h A = {};
      if (inb) {
        union { v16h v; v8h h[2]; } u;
        u.h[0] = *(const v8h*)(xr + cc * 32);
        u.h[1] = *(const v8h*)(xr + cc * 32 + 16);
        A = u.v;
      }
      const _Float16* wf = wtap + ((long long)cc * nNT + nt2 * 2) * 512 + lane * 16;
      v16h B0 = *(const v16h*)(wf);
      v16h B1 = *(const v16h*)(wf + 512);
      acc0 = __builtin_amdgcn_wmma_f32_16x16x32_f16(false, A, false, B0, (short)0, acc0, false, false);
      acc1 = __builtin_amdgcn_wmma_f32_16x16x32_f16(false, A, false, B1, (short)0, acc1, false, false);
    }
  }
  int co0 = nt2 * 32 + (lane & 15);
  int co1 = co0 + 16;
  for (int el = 0; el < 8; ++el) {
    long long pix = mt * 16 + el + 8 * g;
    if (pix < M) {
      int t2 = (int)(pix % L); int b2 = (int)(pix / L);
      _Float16* yp = Yp + (long long)b2 * sB + (long long)t2 * sT;
      float r0 = acc0[el]; if (r0 < 0.f) r0 = 0.f;
      float r1 = acc1[el]; if (r1 < 0.f) r1 = 0.f;
      yp[co0] = (_Float16)((float)yp[co0] + r0);
      yp[co1] = (_Float16)((float)yp[co1] + r1);
    }
  }
}

// ---------------------------------------------------------------------------
// 2x2 max-pool, NHWC f16.
// ---------------------------------------------------------------------------
__global__ void maxpool2_nhwc(const _Float16* __restrict__ x, _Float16* __restrict__ y,
                              int Bn, int H, int W, int C) {
  long long total = (long long)Bn * (H / 2) * (W / 2) * C;
  long long i = (long long)blockIdx.x * 256 + threadIdx.x;
  if (i >= total) return;
  int c = (int)(i % C); long long t = i / C;
  int ox = (int)(t % (W / 2)); t /= (W / 2);
  int oy = (int)(t % (H / 2)); int b = (int)(t / (H / 2));
  const _Float16* p = x + (((long long)b * H + oy * 2) * W + ox * 2) * C + c;
  float m = (float)p[0];
  float v1 = (float)p[C];                        if (v1 > m) m = v1;
  float v2 = (float)p[(long long)W * C];         if (v2 > m) m = v2;
  float v3 = (float)p[(long long)W * C + C];     if (v3 > m) m = v3;
  y[i] = (_Float16)m;
}

// ---------------------------------------------------------------------------
// Final 1x1 conv 128->5 + bias: NHWC f16 -> NCHW fp32 [4,5,36,100].
// ---------------------------------------------------------------------------
__global__ void conv8_head(const _Float16* __restrict__ x, const float* __restrict__ w,
                           const float* __restrict__ b, float* __restrict__ y) {
  int i = blockIdx.x * 256 + threadIdx.x;
  if (i >= 4 * 5 * 3600) return;
  int hw = i % 3600; int t = i / 3600;
  int co = t % 5; int bb = t / 5;
  const _Float16* xp = x + ((long long)bb * 3600 + hw) * 128;
  const float* wp = w + co * 128;
  float acc = 0.f;
  for (int c = 0; c < 128; ++c) acc += (float)xp[c] * wp[c];
  y[i] = acc + b[co];
}

// ---------------------------------------------------------------------------
// Bilinear resize (align_corners) 36x100 -> 288x800, then softmax over C=5.
// ---------------------------------------------------------------------------
__global__ void resize_softmax(const float* __restrict__ x8, float* __restrict__ out) {
  long long i = (long long)blockIdx.x * 256 + threadIdx.x;
  if (i >= 4LL * 288 * 800) return;
  int ox = (int)(i % 800); long long t = i / 800;
  int oy = (int)(t % 288); int bb = (int)(t / 288);
  float fy = oy * (35.0f / 287.0f);
  float fx = ox * (99.0f / 799.0f);
  int y0 = (int)floorf(fy); int y1 = y0 + 1; if (y1 > 35) y1 = 35;
  int x0 = (int)floorf(fx); int x1 = x0 + 1; if (x1 > 99) x1 = 99;
  float wy = fy - (float)y0, wx = fx - (float)x0;
  float v[5]; float mx = -1e30f;
  for (int c = 0; c < 5; ++c) {
    const float* p = x8 + (long long)(bb * 5 + c) * 3600;
    float a  = p[y0 * 100 + x0] * (1.f - wy) + p[y1 * 100 + x0] * wy;
    float bq = p[y0 * 100 + x1] * (1.f - wy) + p[y1 * 100 + x1] * wy;
    float r = a * (1.f - wx) + bq * wx;
    v[c] = r; if (r > mx) mx = r;
  }
  float s = 0.f;
  for (int c = 0; c < 5; ++c) { v[c] = expf(v[c] - mx); s += v[c]; }
  float inv = 1.f / s;
  for (int c = 0; c < 5; ++c)
    out[((long long)(bb * 5 + c) * 288 + oy) * 800 + ox] = v[c] * inv;
}

// ---------------------------------------------------------------------------
// Head 2: softmax over C at each pixel, 2x2 mean-pool -> p2 [4,4500].
// ---------------------------------------------------------------------------
__global__ void softmax_pool(const float* __restrict__ x8, float* __restrict__ p2) {
  int i = blockIdx.x * 256 + threadIdx.x;
  if (i >= 4 * 18 * 50) return;
  int pw = i % 50; int t = i / 50;
  int ph = t % 18; int bb = t / 18;
  float acc[5] = {0.f, 0.f, 0.f, 0.f, 0.f};
  for (int dy = 0; dy < 2; ++dy)
    for (int dx = 0; dx < 2; ++dx) {
      int yy = ph * 2 + dy, xx = pw * 2 + dx;
      float v[5]; float mx = -1e30f;
      for (int c = 0; c < 5; ++c) {
        v[c] = x8[((long long)(bb * 5 + c) * 36 + yy) * 100 + xx];
        if (v[c] > mx) mx = v[c];
      }
      float s = 0.f;
      for (int c = 0; c < 5; ++c) { v[c] = expf(v[c] - mx); s += v[c]; }
      float inv = 1.f / s;
      for (int c = 0; c < 5; ++c) acc[c] += v[c] * inv;
    }
  for (int c = 0; c < 5; ++c)
    p2[bb * 4500 + c * 900 + ph * 50 + pw] = acc[c] * 0.25f;
}

// fc9: one wave per output (b,j); lane-strided dot over 4500 + shfl reduce.
__global__ void fc9_k(const float* __restrict__ p2, const float* __restrict__ w,
                      const float* __restrict__ b, float* __restrict__ h) {
  int wave = blockIdx.x * 8 + (threadIdx.x >> 5);
  int lane = threadIdx.x & 31;
  if (wave >= 4 * 128) return;
  int j = wave % 128, bb = wave / 128;
  const float* xp = p2 + bb * 4500;
  const float* wp = w + (long long)j * 4500;
  float acc = 0.f;
  for (int k = lane; k < 4500; k += 32) acc += xp[k] * wp[k];
  for (int off = 16; off > 0; off >>= 1) acc += __shfl_down(acc, off, 32);
  if (lane == 0) {
    float r = acc + b[j];
    h[wave] = r > 0.f ? r : 0.f;
  }
}

__global__ void fc10_k(const float* __restrict__ h, const float* __restrict__ w,
                       const float* __restrict__ b, float* __restrict__ out) {
  int i = threadIdx.x;
  if (i >= 16) return;
  int j = i % 4, bb = i / 4;
  const float* xp = h + bb * 128;
  const float* wp = w + j * 128;
  float acc = 0.f;
  for (int k = 0; k < 128; ++k) acc += xp[k] * wp[k];
  out[i] = 1.0f / (1.0f + expf(-(acc + b[j])));
}

// ---------------------------------------------------------------------------
// Host orchestration.
// Input dict order: 0:x, then (w,s,b) for 11,12,21,22,31,32,33,41,42,43,51,52,53
// (indices 1..39), 40-42: w6,s6,b6, 43-45: w7,s7,b7, 46:wd 47:wu 48:wr 49:wl,
// 50:w8 51:b8 52:fw9 53:fb9 54:fw10 55:fb10.
// Workspace layout (bytes): [0) actA 117,964,800 | actB 117,964,800 |
//   packed weights 40,296,448 | x8 288,000 | p2 72,000 | h9 2,048 ]  ~277 MB.
// ---------------------------------------------------------------------------
extern "C" void kernel_launch(void* const* d_in, const int* in_sizes, int n_in,
                              void* d_out, int out_size, void* d_ws, size_t ws_size,
                              hipStream_t stream) {
  (void)in_sizes; (void)n_in; (void)out_size; (void)ws_size;
  char* ws = (char*)d_ws;
  _Float16* actA = (_Float16*)ws;
  _Float16* actB = (_Float16*)(ws + 117964800LL);
  _Float16* wpk  = (_Float16*)(ws + 235929600LL);
  float* x8 = (float*)(ws + 276226048LL);
  float* p2 = (float*)(ws + 276514048LL);
  float* h9 = (float*)(ws + 276586048LL);
  float* out = (float*)d_out;
  auto F = [&](int i) { return (const float*)d_in[i]; };

  // ---- repack all WMMA weights (f32 OIHW -> f16 fragment layout) ----
  struct RP { int widx, Cin, Cout, KH, KW; };
  const RP rps[18] = {
    {4, 64, 64, 3, 3},   {7, 64, 128, 3, 3},  {10, 128, 128, 3, 3}, {13, 128, 256, 3, 3},
    {16, 256, 256, 3, 3},{19, 256, 256, 3, 3},{22, 256, 512, 3, 3}, {25, 512, 512, 3, 3},
    {28, 512, 512, 3, 3},{31, 512, 512, 3, 3},{34, 512, 512, 3, 3}, {37, 512, 512, 3, 3},
    {40, 512, 1024, 3, 3},{43, 1024, 128, 1, 1},
    {46, 128, 128, 1, 9},{47, 128, 128, 1, 9},{48, 128, 128, 9, 1},{49, 128, 128, 9, 1}};
  long long woff[18]; long long wacc = 0;
  for (int i = 0; i < 18; ++i) {
    woff[i] = wacc;
    long long tot = (long long)rps[i].KH * rps[i].KW * rps[i].Cin * rps[i].Cout;
    wacc += tot;
    repack_weights<<<(unsigned)DIV_UP(tot, 256), 256, 0, stream>>>(
        F(rps[i].widx), wpk + woff[i], rps[i].Cin, rps[i].Cout, rps[i].KH, rps[i].KW);
  }

  auto conv = [&](const _Float16* Xp, _Float16* Yp, int li, int sIdx, int bIdx,
                  int H, int Wd, int Cin, int Cout, int KH, int KW, int dil, int pad) {
    long long M = 4LL * H * Wd;
    long long waves = ((M + 31) / 32) * (Cout / 32);
    conv_wmma<<<(unsigned)((waves + 7) / 8), 256, 0, stream>>>(
        Xp, wpk + woff[li], F(sIdx), F(bIdx), Yp, 4, H, Wd, Cin, Cout, KH, KW, dil, pad);
  };
  auto pool = [&](const _Float16* Xp, _Float16* Yp, int H, int W, int C) {
    long long tot = 4LL * (H / 2) * (W / 2) * C;
    maxpool2_nhwc<<<(unsigned)DIV_UP(tot, 256), 256, 0, stream>>>(Xp, Yp, 4, H, W, C);
  };

  // ---- backbone ----
  conv1_direct<<<(unsigned)DIV_UP(4LL * 288 * 800 * 64, 256), 256, 0, stream>>>(
      F(0), F(1), F(2), F(3), actA);                               // L11 -> A
  conv(actA, actB, 0,  5,  6, 288, 800,  64,  64, 3, 3, 1, 1);     // L12 -> B
  pool(actB, actA, 288, 800, 64);                                  // -> A [144,400,64]
  conv(actA, actB, 1,  8,  9, 144, 400,  64, 128, 3, 3, 1, 1);     // L21 -> B
  conv(actB, actA, 2, 11, 12, 144, 400, 128, 128, 3, 3, 1, 1);     // L22 -> A
  pool(actA, actB, 144, 400, 128);                                 // -> B [72,200,128]
  conv(actB, actA, 3, 14, 15,  72, 200, 128, 256, 3, 3, 1, 1);     // L31 -> A
  conv(actA, actB, 4, 17, 18,  72, 200, 256, 256, 3, 3, 1, 1);     // L32 -> B
  conv(actB, actA, 5, 20, 21,  72, 200, 256, 256, 3, 3, 1, 1);     // L33 -> A
  pool(actA, actB, 72, 200, 256);                                  // -> B [36,100,256]
  conv(actB, actA, 6, 23, 24,  36, 100, 256, 512, 3, 3, 1, 1);     // L41 -> A
  conv(actA, actB, 7, 26, 27,  36, 100, 512, 512, 3, 3, 1, 1);     // L42 -> B
  conv(actB, actA, 8, 29, 30,  36, 100, 512, 512, 3, 3, 1, 1);     // L43 -> A
  conv(actA, actB, 9, 32, 33,  36, 100, 512, 512, 3, 3, 2, 2);     // L51 -> B
  conv(actB, actA, 10, 35, 36, 36, 100, 512, 512, 3, 3, 2, 2);     // L52 -> A
  conv(actA, actB, 11, 38, 39, 36, 100, 512, 512, 3, 3, 2, 2);     // L53 -> B
  conv(actB, actA, 12, 41, 42, 36, 100, 512, 1024, 3, 3, 4, 4);    // L6  -> A
  conv(actA, actB, 13, 44, 45, 36, 100, 1024, 128, 1, 1, 1, 0);    // L7  -> B

  // ---- SCNN message passing (in place on xs = actB [4,36,100,128] NHWC) ----
  _Float16* xs = actB;
  const long long sBat = 36LL * 100 * 128;
  auto scan = [&](const _Float16* Xp, _Float16* Yp, int li, int L, long long sT) {
    long long waves = ((4LL * L + 15) / 16) * 4;
    scan_step_wmma<<<(unsigned)((waves + 7) / 8), 256, 0, stream>>>(
        Xp, Yp, wpk + woff[li], 4, L, sBat, sT, 128);
  };
  for (int i = 1; i < 36; ++i)                                     // down (wd)
    scan(xs + (long long)(i - 1) * 100 * 128, xs + (long long)i * 100 * 128, 14, 100, 128);
  for (int i = 34; i >= 1; --i)                                    // up (wu)
    scan(xs + (long long)(i + 1) * 100 * 128, xs + (long long)i * 100 * 128, 15, 100, 128);
  for (int j = 1; j < 100; ++j)                                    // right (wr)
    scan(xs + (long long)(j - 1) * 128, xs + (long long)j * 128, 16, 36, 100LL * 128);
  for (int j = 98; j >= 1; --j)                                    // left (wl)
    scan(xs + (long long)(j + 1) * 128, xs + (long long)j * 128, 17, 36, 100LL * 128);

  // ---- heads ----
  conv8_head<<<(unsigned)DIV_UP(4 * 5 * 3600, 256), 256, 0, stream>>>(xs, F(50), F(51), x8);
  resize_softmax<<<(unsigned)DIV_UP(4LL * 288 * 800, 256), 256, 0, stream>>>(x8, out);
  softmax_pool<<<(unsigned)DIV_UP(4 * 18 * 50, 256), 256, 0, stream>>>(x8, p2);
  fc9_k<<<64, 256, 0, stream>>>(p2, F(52), F(53), h9);
  fc10_k<<<1, 32, 0, stream>>>(h9, F(54), F(55), out + 4608000);
}
